// ISTFT_86053964743182
// MI455X (gfx1250) — matrix operand order repaired and checked by reading, not compiled, not run on hardware
//
#include <hip/hip_runtime.h>

// ---------------------------------------------------------------------------
// ISTFT = bf16 WMMA GEMM  A[32000,512] x W[512,512]  (Nyquist bin folded into
// the epilogue as a rank-1 update) + fused overlap-add via f32 atomics.
// B tiles staged HBM->LDS with gfx1250 global_load_async_to_lds_b128.
// ---------------------------------------------------------------------------

typedef __attribute__((ext_vector_type(16))) __bf16 v16bf;
typedef __attribute__((ext_vector_type(8)))  float  v8f;

#define KD_ROW   514            // physical row stride of A (2*257 floats)
#define KD       512            // GEMM reduction dim (f = 0..255 re/im)
#define NWIN     512            // frame length (N dim)
#define STRIDE_T 128
#define BATCH    16
#define TFRAMES  2000
#define MROWS    (BATCH * TFRAMES)                 // 32000
#define LOUT     ((TFRAMES - 1) * STRIDE_T + NWIN) // 256384

__device__ __forceinline__ unsigned short f2bf(float f) {
    union { float f; unsigned u; } v; v.f = f;
    unsigned u = v.u;
    u += 0x7FFFu + ((u >> 16) & 1u);     // round-to-nearest-even
    return (unsigned short)(u >> 16);
}

// gfx1250 async copy: 16B global -> LDS, tracked by ASYNCcnt (ISA 15.18.3 #98)
__device__ __forceinline__ void async_ldst_b128(unsigned lds_off, const void* g) {
    asm volatile("global_load_async_to_lds_b128 %0, %1, off"
                 :: "v"(lds_off), "v"((unsigned long long)(size_t)g)
                 : "memory");
}
__device__ __forceinline__ void wait_async0() {
    asm volatile("s_wait_asynccnt 0x0" ::: "memory");
}

// ---- zero the output (harness poisons d_out) ------------------------------
__global__ void istft_zero(float* __restrict__ out, int n) {
    int i  = blockIdx.x * blockDim.x + threadIdx.x;
    int st = gridDim.x * blockDim.x;
    for (; i < n; i += st) out[i] = 0.0f;
}

// ---- W_t[NWIN][KD] bf16: W_t[n][2f] = cos[f][n], W_t[n][2f+1] = -sin[f][n] -
__global__ void istft_build_w(const float* __restrict__ ccos,
                              const float* __restrict__ csin,
                              unsigned short* __restrict__ wt) {
    int i  = blockIdx.x * blockDim.x + threadIdx.x;
    int st = gridDim.x * blockDim.x;
    for (; i < NWIN * KD; i += st) {
        int n = i >> 9;          // / 512
        int k = i & 511;
        int f = k >> 1;
        float v = (k & 1) ? -csin[f * NWIN + n] : ccos[f * NWIN + n];
        wt[i] = f2bf(v);
    }
}

// ---- main GEMM + overlap-add ----------------------------------------------
__global__ __launch_bounds__(256)
void istft_gemm(const float* __restrict__ A,            // [MROWS][KD_ROW] f32
                const unsigned short* __restrict__ Wt,  // [NWIN][KD] bf16
                const float* __restrict__ cnyq,         // coef_cos row f=256
                float* __restrict__ out) {              // [BATCH][LOUT]
    __shared__ __align__(16) unsigned short lA[2][128 * 32];   //  8 KB each
    __shared__ __align__(16) unsigned short lB[2][256 * 32];   // 16 KB each

    const int tid  = threadIdx.x;
    const int lane = tid & 31;
    const int wid  = tid >> 5;
    const int mw   = wid & 1;                 // 2 wave rows  (64 M each)
    const int nw   = wid >> 1;                // 4 wave cols  (64 N each)
    const int m0   = blockIdx.x * 128;
    const int n0   = blockIdx.y * 256;

    v8f acc[4][4] = {};                       // 4 M-tiles x 4 N-tiles (16x16)

    // staging assignments
    const int ar = tid >> 2;                  // A: 64 rows / pass, 2 passes
    const int ac = (tid & 3) * 8;             // A: 8 cols per thread

    auto stage = [&](int ks, int buf) {
        const int c0 = ks * 32;
        // B tile 256(n) x 32(k) bf16: 4 async 16B copies straight to LDS
        {
            const unsigned short* src = &Wt[(long)(n0 + tid) * KD + c0];
            unsigned dst = (unsigned)(size_t)&lB[buf][tid * 32];
#pragma unroll
            for (int j = 0; j < 4; ++j)
                async_ldst_b128(dst + j * 16, src + j * 8);
        }
        // A tile 128 x 32: f32 -> bf16 through VGPRs
#pragma unroll
        for (int p = 0; p < 2; ++p) {
            int r = ar + p * 64;
            const float* src = &A[(long)(m0 + r) * KD_ROW + c0 + ac];
            float x[8];
#pragma unroll
            for (int q = 0; q < 4; ++q) {
                float2 t = *(const float2*)(src + 2 * q);
                x[2 * q] = t.x; x[2 * q + 1] = t.y;
            }
            unsigned pk[4];
#pragma unroll
            for (int q = 0; q < 4; ++q)
                pk[q] = (unsigned)f2bf(x[2 * q]) | ((unsigned)f2bf(x[2 * q + 1]) << 16);
            *(uint4*)&lA[buf][r * 32 + ac] = make_uint4(pk[0], pk[1], pk[2], pk[3]);
        }
    };

    stage(0, 0);

    // fragment addressing per CDNA5 WMMA VGPR layouts (ISA 7.12.2)
    const int aRow = lane & 15;
    const int aKb  = (lane < 16) ? 0 : 8;     // A: K {0..7,16..23} / {8..15,24..31}
    const int bN   = lane & 15;
    const int bKb  = (lane < 16) ? 0 : 8;     // B: K {0..7,16..23} / {8..15,24..31}

    union Frag { uint4 q[2]; v16bf v; };

    for (int ks = 0; ks < 16; ++ks) {
        wait_async0();                        // our async B copies done
        __syncthreads();                      // tile[buf] visible block-wide
        const int buf = ks & 1;
        if (ks + 2 < 16)
            __builtin_prefetch(A + (long)(m0 + ar) * KD_ROW + (ks + 2) * 32, 0, 0);
        if (ks < 15) stage(ks + 1, buf ^ 1);

        Frag af[4];
#pragma unroll
        for (int mt = 0; mt < 4; ++mt) {
            int row = mw * 64 + mt * 16 + aRow;
            af[mt].q[0] = *(const uint4*)&lA[buf][row * 32 + aKb];
            af[mt].q[1] = *(const uint4*)&lA[buf][row * 32 + aKb + 16];
        }
#pragma unroll
        for (int nt = 0; nt < 4; ++nt) {
            int n = nw * 64 + nt * 16 + bN;
            Frag bfr;
            bfr.q[0] = *(const uint4*)&lB[buf][n * 32 + bKb];
            bfr.q[1] = *(const uint4*)&lB[buf][n * 32 + bKb + 16];
#pragma unroll
            for (int mt = 0; mt < 4; ++mt) {
                acc[mt][nt] = __builtin_amdgcn_wmma_f32_16x16x32_bf16(
                    false, af[mt].v, false, bfr.v,
                    (short)0, acc[mt][nt], false, false);
            }
        }
    }

    // epilogue: fold Nyquist rank-1 term, then overlap-add with f32 atomics.
    // C layout: VGPR r, lane L -> row = r + 8*(L>=16), col = L%16
    const int rowOff = ((lane >> 4) << 3);
    const int colOff = lane & 15;
#pragma unroll
    for (int mt = 0; mt < 4; ++mt) {
        int rowBase = m0 + mw * 64 + mt * 16 + rowOff;
        float anyq[8];
#pragma unroll
        for (int r = 0; r < 8; ++r)
            anyq[r] = A[(long)(rowBase + r) * KD_ROW + 512];   // re @ f=256
#pragma unroll
        for (int nt = 0; nt < 4; ++nt) {
            int col = n0 + nw * 64 + nt * 16 + colOff;
            float wn = cnyq[col];
#pragma unroll
            for (int r = 0; r < 8; ++r) {
                int gm = rowBase + r;
                int b  = gm / TFRAMES;
                int t  = gm - b * TFRAMES;
                atomicAdd(&out[(long)b * LOUT + t * STRIDE_T + col],
                          acc[mt][nt][r] + anyq[r] * wn);
            }
        }
    }
}

// ---------------------------------------------------------------------------
extern "C" void kernel_launch(void* const* d_in, const int* in_sizes, int n_in,
                              void* d_out, int out_size, void* d_ws, size_t ws_size,
                              hipStream_t stream) {
    const float* x    = (const float*)d_in[0];   // [16,2000,257,2]
    const float* ccos = (const float*)d_in[1];   // [257,512]
    const float* csin = (const float*)d_in[2];   // [257,512]
    float* out        = (float*)d_out;           // [16,1,256384]
    unsigned short* wt = (unsigned short*)d_ws;  // 512*512*2 = 512 KB

    istft_zero<<<2048, 256, 0, stream>>>(out, out_size);
    istft_build_w<<<(NWIN * KD + 255) / 256, 256, 0, stream>>>(ccos, csin, wt);

    dim3 grid(MROWS / 128, NWIN / 256);          // 250 x 2
    istft_gemm<<<grid, 256, 0, stream>>>(x, wt, ccos + 256 * NWIN, out);
}
